// AllObsPredictAtten_35880156791540
// MI455X (gfx1250) — compile-verified, matrix operand order checked by reading, not run on hardware
//
#include <hip/hip_runtime.h>
#include <math.h>

typedef __attribute__((ext_vector_type(16))) __bf16 v16bf;
typedef __attribute__((ext_vector_type(8)))  float  v8f;

#define NT 256

// ---- dynamic LDS layout (byte offsets, all 16B aligned) ----
#define OFF_BIG    0        // 256000 B: grid A / W_comb / inv A / module W / sMO
#define OFF_SW     256000   // 20480 B : stage weight, bf16 [32][KP<=320]
#define OFF_SC     276480   // 25600 B : stage A (grid_comb / c3 / goal / h)
#define OFF_ONE    302080   // 6400 B  : grid_one bf16 [400][8]
#define OFF_WONE   308480   // 1024 B  : W_onehot f32 [32][8]
#define OFF_BEMB   309504   // 128 B
#define OFF_BONE   309632   // 128 B
#define OFF_BINV1  309760   // 128 B
#define OFF_INB    309888   // 1536 B  : in_b f32 [12][32]
#define OFF_OUTB   311424   // 1536 B  : out_b f32 [12][32]
#define OFF_ATTW   312960   // 1536 B  : att_W f32 [12][32]
#define OFF_ATTB   314496   // 64 B
#define OFF_NET    314560   // 3072 B  : net bf16 [16][3][32]
#define OFF_ACC    317632   // 2048 B  : f32 [16][32] reduction buffer
#define OFF_LOGIT  319680   // 768 B
#define OFF_SEL    320448   // 768 B
#define SMEM_BYTES 321216

__device__ __forceinline__ unsigned pack2(float a, float b) {
  union { __bf16 h[2]; unsigned u; } w;
  w.h[0] = (__bf16)a; w.h[1] = (__bf16)b;
  return w.u;
}

// Load one 16x32 bf16 fragment (A layout; B uses same loader on [N][K] weights).
// lanes 0-15: rows 0-15, K 0..7 and 16..23 ; lanes 16-31: same rows, K +8.
__device__ __forceinline__ v16bf ld_frag(const __bf16* p0, int stride) {
  const int lane = threadIdx.x & 31;
  const int rr   = lane & 15;
  const int hh   = (lane >> 4) & 1;
  const __bf16* p = p0 + rr * stride + hh * 8;
  union { v16bf v; uint4 q[2]; } u;
  u.q[0] = *(const uint4*)(p);
  u.q[1] = *(const uint4*)(p + 16);
  return u.v;
}

__device__ __forceinline__ v8f wmma_bf16(v16bf a, v16bf b, v8f c) {
  return __builtin_amdgcn_wmma_f32_16x16x32_bf16(false, a, false, b,
                                                 (short)0, c, false, false);
}

// convert + zero-pad a (32 x K) f32 weight into LDS bf16 (32 x KP); KP even
__device__ __forceinline__ void load_w(__bf16* dst, const float* W, int K, int KP, int t) {
  const int pairs = 32 * KP / 2;
  for (int e = t; e < pairs; e += NT) {
    int k = (e * 2) % KP;
    int o = (e * 2) / KP;
    float a = (k     < K) ? W[o * K + k]     : 0.f;
    float b = (k + 1 < K) ? W[o * K + k + 1] : 0.f;
    ((unsigned*)dst)[e] = pack2(a, b);
  }
}

extern "C" __global__ __launch_bounds__(NT)
void fused_allobs(const float* __restrict__ x,
                  const float* __restrict__ hiddens,
                  const float* __restrict__ Wemb,  const float* __restrict__ bemb,
                  const float* __restrict__ Wone,  const float* __restrict__ bone,
                  const float* __restrict__ Wcomb, const float* __restrict__ bcomb,
                  const float* __restrict__ Winv1, const float* __restrict__ binv1,
                  const float* __restrict__ Winv2, const float* __restrict__ binv2,
                  const float* __restrict__ Wgoal, const float* __restrict__ bgoal,
                  const float* __restrict__ inW,   const float* __restrict__ inB,
                  const float* __restrict__ outW,  const float* __restrict__ outB,
                  const float* __restrict__ attW,  const float* __restrict__ attB,
                  float* __restrict__ out) {
  extern __shared__ char smem[];
  const int t    = threadIdx.x;
  const int wid  = t >> 5;
  const int lane = t & 31;
  const int rl   = lane & 15;
  const int hh   = (lane >> 4) & 1;
  const int row0 = blockIdx.x * 16;

  __bf16* sBig  = (__bf16*)(smem + OFF_BIG);
  __bf16* sW    = (__bf16*)(smem + OFF_SW);
  __bf16* sC    = (__bf16*)(smem + OFF_SC);
  __bf16* sOne  = (__bf16*)(smem + OFF_ONE);
  float*  sWone = (float*)(smem + OFF_WONE);
  float*  sBemb = (float*)(smem + OFF_BEMB);
  float*  sBone = (float*)(smem + OFF_BONE);
  float*  sBinv1= (float*)(smem + OFF_BINV1);
  float*  sInB  = (float*)(smem + OFF_INB);
  float*  sOutB = (float*)(smem + OFF_OUTB);
  float*  sAttW = (float*)(smem + OFF_ATTW);
  float*  sAttB = (float*)(smem + OFF_ATTB);
  __bf16* sNet  = (__bf16*)(smem + OFF_NET);
  float*  sAcc  = (float*)(smem + OFF_ACC);
  float*  sLogit= (float*)(smem + OFF_LOGIT);
  float*  sSel  = (float*)(smem + OFF_SEL);
  __bf16* sWin  = sBig;            // bf16 in_W  [12][32][32]
  __bf16* sWout = sBig + 12288;    // bf16 out_W [12][32][32]
  float*  sMO   = (float*)(smem + OFF_BIG + 49152); // f32 [12][16][32]
  __bf16* sH    = sC;              // bf16 h [12][16][32]

  const v8f vzero = {0.f,0.f,0.f,0.f,0.f,0.f,0.f,0.f};

  // ---- Phase 0: stage grid_emb A (400x320 bf16), grid_one, W_embed, biases, att ----
  for (int e = t; e < 64000; e += NT) {          // 400 pos-rows x 160 pairs
    int pr = e / 160;
    int k  = (e - pr * 160) * 2;
    int br = pr / 25, p = pr - br * 25;
    float a = 0.f, b = 0.f;
    if (k < 300) {
      const float* g = x + (size_t)(row0 + br) * 10975 + p * 300 + k;
      a = g[0]; b = g[1];
    }
    ((unsigned*)sBig)[pr * 160 + (k >> 1)] = pack2(a, b);
  }
  for (int e = t; e < 3200; e += NT) {           // grid_one 400x8 (pad 7->8)
    int pr = e >> 3, j = e & 7;
    float v = 0.f;
    if (j < 7) {
      int br = pr / 25, p = pr - br * 25;
      v = x[(size_t)(row0 + br) * 10975 + 7500 + p * 7 + j];
    }
    sOne[e] = (__bf16)v;
  }
  for (int e = t; e < 256; e += NT) {            // W_onehot 32x8 f32
    int o = e >> 3, j = e & 7;
    sWone[e] = (j < 7) ? Wone[o * 7 + j] : 0.f;
  }
  if (t < 32) { sBemb[t] = bemb[t]; sBone[t] = bone[t]; }
  for (int e = t; e < 384; e += NT) sAttW[e] = attW[e];
  if (t < 12) sAttB[t] = attB[t];
  load_w(sW, Wemb, 300, 320, t);

  // Warm L2 (global_prefetch_b8) with this block's goal+inventory span of x
  // (contiguous: cols 7675..10975 = 13200 B/row) and hiddens, so HBM stays
  // busy while the 500 grid-GEMM WMMAs below run out of LDS.
  for (int e = t; e < 16 * 104; e += NT) {
    int br = e / 104, c = e - br * 104;
    __builtin_prefetch(x + (size_t)(row0 + br) * 10975 + 7675 + c * 32, 0, 1);
  }
  if (t < 16) __builtin_prefetch(hiddens + (size_t)(row0 + t) * 32, 0, 1);
  __syncthreads();

  // ---- Phase 1: c1 GEMM (400x320)@(320x32) + c2 epilogue -> grid_comb sC[16][800] ----
  for (int j = wid; j < 50; j += 8) {
    int mt = j >> 1, nt = j & 1;
    const __bf16* abase = sBig + mt * 16 * 320;
    const __bf16* bbase = sW   + nt * 16 * 320;
    v8f acc = vzero;
    v16bf a = ld_frag(abase, 320);
    v16bf b = ld_frag(bbase, 320);
#pragma unroll
    for (int kc = 0; kc < 10; ++kc) {            // software-pipelined K loop
      v16bf an = a, bn = b;
      if (kc < 9) {
        an = ld_frag(abase + (kc + 1) * 32, 320);
        bn = ld_frag(bbase + (kc + 1) * 32, 320);
      }
      acc = wmma_bf16(a, b, acc);
      a = an; b = bn;
    }
    union { v8f v; float f[8]; } ua; ua.v = acc;
    int n = nt * 16 + rl;
#pragma unroll
    for (int i = 0; i < 8; ++i) {
      int m  = mt * 16 + i + hh * 8;
      int br = m / 25, p = m - br * 25;
      float v1 = fmaxf(ua.f[i] + sBemb[n], 0.f);
      float v2 = sBone[n];
#pragma unroll
      for (int jj = 0; jj < 7; ++jj)
        v2 += (float)sOne[m * 8 + jj] * sWone[n * 8 + jj];
      v2 = fmaxf(v2, 0.f);
      sC[br * 800 + p * 32 + n] = (__bf16)(v1 + v2);
    }
  }
  __syncthreads();

  // ---- Phase 2: W_comb -> sBig (bf16 32x800), init sAcc = b_comb ----
  load_w(sBig, Wcomb, 800, 800, t);
  for (int e = t; e < 512; e += NT) sAcc[e] = bcomb[e & 31];
  __syncthreads();

  // ---- Phase 3: grid_comb GEMM (16x800)@(800x32), K split over waves ----
  {
    int nt = wid & 1, ks = wid >> 1;   // 8 waves = 2 ntiles x 4 K-slices
    const __bf16* bbase = sBig + nt * 16 * 800;
    v8f acc = vzero;
    v16bf a = ld_frag(sC    + ks * 32, 800);
    v16bf b = ld_frag(bbase + ks * 32, 800);
    for (int kc = ks; kc < 25; kc += 4) {
      int kn = kc + 4;
      v16bf an = a, bn = b;
      if (kn < 25) {
        an = ld_frag(sC    + kn * 32, 800);
        bn = ld_frag(bbase + kn * 32, 800);
      }
      acc = wmma_bf16(a, b, acc);
      a = an; b = bn;
    }
    union { v8f v; float f[8]; } ua; ua.v = acc;
    int n = nt * 16 + rl;
#pragma unroll
    for (int i = 0; i < 8; ++i)
      atomicAdd(&sAcc[(i + hh * 8) * 32 + n], ua.f[i]);
  }
  __syncthreads();

  // ---- Phase 4: net[:,0] = relu(sAcc); stage inventory A (160x320) + W_inv1 ----
  for (int e = t; e < 512; e += NT)
    sNet[((e >> 5) * 3 + 0) * 32 + (e & 31)] = (__bf16)fmaxf(sAcc[e], 0.f);
  for (int e = t; e < 25600; e += NT) {          // 160 rows x 160 pairs
    int pr = e / 160;
    int k  = (e - pr * 160) * 2;
    int br = pr / 10, ip = pr - br * 10;
    float a = 0.f, b = 0.f;
    if (k < 300) {
      const float* g = x + (size_t)(row0 + br) * 10975 + 7975 + ip * 300 + k;
      a = g[0]; b = g[1];
    }
    ((unsigned*)sBig)[pr * 160 + (k >> 1)] = pack2(a, b);
  }
  load_w(sW, Winv1, 300, 320, t);
  if (t < 32) sBinv1[t] = binv1[t];
  __syncthreads();

  // ---- Phase 5: c3 GEMM (160x320)@(320x32) -> sC as c3 [16][320] ----
  for (int j = wid; j < 20; j += 8) {
    int mt = j >> 1, nt = j & 1;
    const __bf16* abase = sBig + mt * 16 * 320;
    const __bf16* bbase = sW   + nt * 16 * 320;
    v8f acc = vzero;
    v16bf a = ld_frag(abase, 320);
    v16bf b = ld_frag(bbase, 320);
#pragma unroll
    for (int kc = 0; kc < 10; ++kc) {
      v16bf an = a, bn = b;
      if (kc < 9) {
        an = ld_frag(abase + (kc + 1) * 32, 320);
        bn = ld_frag(bbase + (kc + 1) * 32, 320);
      }
      acc = wmma_bf16(a, b, acc);
      a = an; b = bn;
    }
    union { v8f v; float f[8]; } ua; ua.v = acc;
    int n = nt * 16 + rl;
#pragma unroll
    for (int i = 0; i < 8; ++i) {
      int m  = mt * 16 + i + hh * 8;       // 0..159
      int br = m / 10, ip = m - br * 10;
      sC[br * 320 + ip * 32 + n] = (__bf16)fmaxf(ua.f[i] + sBinv1[n], 0.f);
    }
  }
  __syncthreads();

  // ---- Phase 6: W_inv2 -> sW, init sAcc = b_inv2 ----
  load_w(sW, Winv2, 320, 320, t);
  for (int e = t; e < 512; e += NT) sAcc[e] = binv2[e & 31];
  __syncthreads();

  // ---- Phase 7: inv_embed GEMM (16x320)@(320x32) ----
  {
    int nt = wid & 1, ks = wid >> 1;
    const __bf16* bbase = sW + nt * 16 * 320;
    v8f acc = vzero;
    v16bf a = ld_frag(sC    + ks * 32, 320);
    v16bf b = ld_frag(bbase + ks * 32, 320);
    for (int kc = ks; kc < 10; kc += 4) {
      int kn = kc + 4;
      v16bf an = a, bn = b;
      if (kn < 10) {
        an = ld_frag(sC    + kn * 32, 320);
        bn = ld_frag(bbase + kn * 32, 320);
      }
      acc = wmma_bf16(a, b, acc);
      a = an; b = bn;
    }
    union { v8f v; float f[8]; } ua; ua.v = acc;
    int n = nt * 16 + rl;
#pragma unroll
    for (int i = 0; i < 8; ++i)
      atomicAdd(&sAcc[(i + hh * 8) * 32 + n], ua.f[i]);
  }
  __syncthreads();

  // ---- Phase 8: net[:,1] = relu(sAcc); stage goal A (16x320) + W_goal ----
  for (int e = t; e < 512; e += NT)
    sNet[((e >> 5) * 3 + 1) * 32 + (e & 31)] = (__bf16)fmaxf(sAcc[e], 0.f);
  for (int e = t; e < 2560; e += NT) {
    int br = e / 160;
    int k  = (e - br * 160) * 2;
    float a = 0.f, b = 0.f;
    if (k < 300) {
      const float* g = x + (size_t)(row0 + br) * 10975 + 7675 + k;
      a = g[0]; b = g[1];
    }
    ((unsigned*)sC)[br * 160 + (k >> 1)] = pack2(a, b);
  }
  load_w(sW, Wgoal, 300, 320, t);
  __syncthreads();

  // ---- Phase 9: init sAcc = b_goal ----
  for (int e = t; e < 512; e += NT) sAcc[e] = bgoal[e & 31];
  __syncthreads();

  // ---- Phase 10: goal_embed GEMM (16x320)@(320x32) ----
  {
    int nt = wid & 1, ks = wid >> 1;
    const __bf16* bbase = sW + nt * 16 * 320;
    v8f acc = vzero;
    v16bf a = ld_frag(sC    + ks * 32, 320);
    v16bf b = ld_frag(bbase + ks * 32, 320);
    for (int kc = ks; kc < 10; kc += 4) {
      int kn = kc + 4;
      v16bf an = a, bn = b;
      if (kn < 10) {
        an = ld_frag(sC    + kn * 32, 320);
        bn = ld_frag(bbase + kn * 32, 320);
      }
      acc = wmma_bf16(a, b, acc);
      a = an; b = bn;
    }
    union { v8f v; float f[8]; } ua; ua.v = acc;
    int n = nt * 16 + rl;
#pragma unroll
    for (int i = 0; i < 8; ++i)
      atomicAdd(&sAcc[(i + hh * 8) * 32 + n], ua.f[i]);
  }
  __syncthreads();

  // ---- Phase 11: net[:,2]; module weights/biases; attention logits ----
  for (int e = t; e < 512; e += NT)
    sNet[((e >> 5) * 3 + 2) * 32 + (e & 31)] = (__bf16)fmaxf(sAcc[e], 0.f);
  for (int e = t; e < 6144; e += NT)
    ((unsigned*)sWin)[e]  = pack2(inW[2 * e],  inW[2 * e + 1]);
  for (int e = t; e < 6144; e += NT)
    ((unsigned*)sWout)[e] = pack2(outW[2 * e], outW[2 * e + 1]);
  for (int e = t; e < 384; e += NT) { sInB[e] = inB[e]; sOutB[e] = outB[e]; }
  if (t < 192) {
    int rr = t / 12, m = t - rr * 12;
    float l = sAttB[m];
    const float* hv = hiddens + (size_t)(row0 + rr) * 32;
#pragma unroll
    for (int k = 0; k < 32; ++k) l += hv[k] * sAttW[m * 32 + k];
    sLogit[rr * 12 + m] = l;
  }
  __syncthreads();

  // ---- Phase 12: h = tanh(net_rep @ in_W[m].T + in_b) -> sH ----
  for (int j = wid; j < 24; j += 8) {
    int m = j >> 1, nt = j & 1, g = m >> 2;     // jnp.repeat: module m uses net[m//4]
    v16bf a = ld_frag(sNet + g * 32, 96);
    v16bf b = ld_frag(sWin + m * 1024 + nt * 16 * 32, 32);
    v8f acc = wmma_bf16(a, b, vzero);
    union { v8f v; float f[8]; } ua; ua.v = acc;
    int n = nt * 16 + rl;
#pragma unroll
    for (int i = 0; i < 8; ++i) {
      int rr = i + hh * 8;
      sH[(m * 16 + rr) * 32 + n] = (__bf16)tanhf(ua.f[i] + sInB[m * 32 + n]);
    }
  }
  __syncthreads();

  // ---- Phase 13: module_outs = h @ out_W[m].T + out_b -> sMO; softmax -> sSel ----
  for (int j = wid; j < 24; j += 8) {
    int m = j >> 1, nt = j & 1;
    v16bf a = ld_frag(sH    + m * 512, 32);
    v16bf b = ld_frag(sWout + m * 1024 + nt * 16 * 32, 32);
    v8f acc = wmma_bf16(a, b, vzero);
    union { v8f v; float f[8]; } ua; ua.v = acc;
    int n = nt * 16 + rl;
#pragma unroll
    for (int i = 0; i < 8; ++i)
      sMO[(m * 16 + (i + hh * 8)) * 32 + n] = ua.f[i] + sOutB[m * 32 + n];
  }
  if (t < 16) {
    float mx = -1e30f;
#pragma unroll
    for (int m = 0; m < 12; ++m) mx = fmaxf(mx, sLogit[t * 12 + m]);
    float ex[12], s = 0.f;
#pragma unroll
    for (int m = 0; m < 12; ++m) { ex[m] = expf(sLogit[t * 12 + m] - mx); s += ex[m]; }
    float inv = 1.f / s;
#pragma unroll
    for (int m = 0; m < 12; ++m) sSel[t * 12 + m] = ex[m] * inv;
  }
  __syncthreads();

  // ---- Phase 14: out[b,o] = sum_m sel[b,m] * module_outs[b,m,o] ----
  for (int e = t; e < 512; e += NT) {
    int rr = e >> 5, n = e & 31;
    float s = 0.f;
#pragma unroll
    for (int m = 0; m < 12; ++m)
      s += sSel[rr * 12 + m] * sMO[(m * 16 + rr) * 32 + n];
    out[(size_t)(row0 + rr) * 32 + n] = s;
  }
}

extern "C" void kernel_launch(void* const* d_in, const int* in_sizes, int n_in,
                              void* d_out, int out_size, void* d_ws, size_t ws_size,
                              hipStream_t stream) {
  (void)n_in; (void)out_size; (void)d_ws; (void)ws_size;
  const float* x      = (const float*)d_in[0];
  const float* hid    = (const float*)d_in[1];
  const float* Wemb   = (const float*)d_in[2];
  const float* bemb   = (const float*)d_in[3];
  const float* Wone   = (const float*)d_in[4];
  const float* bone   = (const float*)d_in[5];
  const float* Wcomb  = (const float*)d_in[6];
  const float* bcomb  = (const float*)d_in[7];
  const float* Winv1  = (const float*)d_in[8];
  const float* binv1  = (const float*)d_in[9];
  const float* Winv2  = (const float*)d_in[10];
  const float* binv2  = (const float*)d_in[11];
  const float* Wgoal  = (const float*)d_in[12];
  const float* bgoal  = (const float*)d_in[13];
  const float* inW    = (const float*)d_in[14];
  const float* inB    = (const float*)d_in[15];
  const float* outW   = (const float*)d_in[16];
  const float* outB   = (const float*)d_in[17];
  const float* attW   = (const float*)d_in[18];
  const float* attB   = (const float*)d_in[19];
  float* out = (float*)d_out;

  int Bsz  = in_sizes[1] / 32;      // hiddens is (B, 32)
  int grid = Bsz / 16;
  if (grid <= 0) return;

  (void)hipFuncSetAttribute((const void*)fused_allobs,
                            hipFuncAttributeMaxDynamicSharedMemorySize, SMEM_BYTES);
  fused_allobs<<<grid, NT, SMEM_BYTES, stream>>>(
      x, hid, Wemb, bemb, Wone, bone, Wcomb, bcomb, Winv1, binv1,
      Winv2, binv2, Wgoal, bgoal, inW, inB, outW, outB, attW, attB, out);
}
